// Model_52518860096303
// MI455X (gfx1250) — matrix operand order repaired and checked by reading, not compiled, not run on hardware
//
#include <hip/hip_runtime.h>
#include <hip/hip_bf16.h>
#include <math.h>

// ---------------------------------------------------------------------------
// Problem constants (from the reference): B=8, S=2048, D=2048
// ---------------------------------------------------------------------------
#define DIM_B 8
#define DIM_S 2048
#define DIM_D 2048
#define DIM_N (DIM_B * DIM_S)   // 16384 tokens
#define DIM_3D (3 * DIM_D)      // 6144

typedef __bf16 bf16;
typedef __attribute__((ext_vector_type(16))) __bf16 v16bf;
typedef __attribute__((ext_vector_type(8)))  __bf16 v8bf;
typedef __attribute__((ext_vector_type(8)))  float  v8f;
typedef __attribute__((ext_vector_type(4)))  int    v4i;

union AFrag { v16bf v; v8bf h[2]; };

// ---------------------------------------------------------------------------
// CDNA5 async global->LDS copy (ASYNCcnt path), with safe fallbacks.
// Builtin signature (probe-verified via hipcc diagnostics): first arg is
// v4i addrspace(1)* source, second is LDS destination, then imm offset+cpol.
// ---------------------------------------------------------------------------
#if __has_builtin(__builtin_amdgcn_global_load_async_to_lds_b128)
  #define HAVE_ASYNC_LDS 1
  typedef __attribute__((address_space(1))) v4i gv4i;
  typedef __attribute__((address_space(3))) v4i lv4i;
  // copies 16B per lane: LDS[ldsDst] = MEM[gSrc]
  #define ASYNC_CP16(ldsDst, gSrc) \
    __builtin_amdgcn_global_load_async_to_lds_b128((gv4i*)(gSrc), (lv4i*)(ldsDst), 0, 0)
#else
  // synchronous fallback: reg -> LDS store (still correct, no ASYNCcnt)
  #define ASYNC_CP16(ldsDst, gSrc) \
    do { *(v8bf*)(ldsDst) = *(const v8bf*)(gSrc); } while (0)
#endif

#ifdef HAVE_ASYNC_LDS
  #if __has_builtin(__builtin_amdgcn_s_wait_asynccnt)
    #define WAIT_ASYNC(n) __builtin_amdgcn_s_wait_asynccnt(n)
  #else
    #define WAIT_ASYNC(n) asm volatile("s_wait_asynccnt %0" :: "i"(n) : "memory")
  #endif
#else
  #define WAIT_ASYNC(n) ((void)0)
#endif

// ---------------------------------------------------------------------------
// fp32 -> bf16 elementwise convert
// ---------------------------------------------------------------------------
__global__ __launch_bounds__(256) void k_cvt_f32_bf16(const float* __restrict__ in,
                                                      bf16* __restrict__ out, int n) {
  int i = blockIdx.x * 256 + threadIdx.x;
  if (i < n) out[i] = (bf16)in[i];
}

// ---------------------------------------------------------------------------
// bf16 tiled transpose: out[c*ldout + r] = in[r*ldin + c]
// grid = (C/32, R/32), block = 256 (32x8)
// ---------------------------------------------------------------------------
__global__ __launch_bounds__(256) void k_transpose_bf16(const bf16* __restrict__ in, int ldin,
                                                        bf16* __restrict__ out, int ldout) {
  __shared__ bf16 tile[32][33];
  const int c0 = blockIdx.x * 32;
  const int r0 = blockIdx.y * 32;
  const int tx = threadIdx.x & 31;
  const int ty = threadIdx.x >> 5;   // 0..7
  #pragma unroll
  for (int i = 0; i < 32; i += 8)
    tile[ty + i][tx] = in[(size_t)(r0 + ty + i) * ldin + c0 + tx];
  __syncthreads();
  #pragma unroll
  for (int i = 0; i < 32; i += 8)
    out[(size_t)(c0 + ty + i) * ldout + r0 + tx] = tile[tx][ty + i];
}

// ---------------------------------------------------------------------------
// GEMM: C[M,Nc] = scale * (A[M,K] @ B[Nc,K]^T) + bias[Nc]
// A, B row-major bf16.  Output fp32 or bf16.
// Workgroup tile 128x128, 8 waves (wave32), wave tile 32x64 (2x4 WMMA accums).
// B operand is double-buffered through LDS via CDNA5 async global->LDS copies
// (4x reuse inside the block); A is read direct from global (2x reuse, L2-hot).
// K-stage = 64 (two 32-wide WMMA k-steps per stage).
// All of M, Nc assumed multiples of 128; K multiple of 64.
// ---------------------------------------------------------------------------
#define KT      64
#define BPITCH  72   // 64 + 8 pad elements -> conflict-free ds_load_b128

template <bool OUT_BF16, bool HAS_BIAS>
__global__ __launch_bounds__(256) void k_gemm_abt(
    const bf16* __restrict__ A, int lda,
    const bf16* __restrict__ B, int ldb,
    const float* __restrict__ bias,
    void* __restrict__ Cout, int ldc,
    int K, float scale)
{
  __shared__ bf16 Bs[2][128 * BPITCH];   // 2 x 18KB = 36KB

  const int tid     = threadIdx.x;
  const int lane    = tid & 31;
  const int wave    = tid >> 5;    // 0..7
  const int waveM   = wave >> 1;   // 0..3
  const int waveN   = wave & 1;    // 0..1
  const int l16     = lane & 15;
  const int halfSel = lane >> 4;   // 0 or 1

  const int mBase  = blockIdx.y * 128 + waveM * 32;
  const int nBase0 = blockIdx.x * 128;
  const int nWave  = waveN * 64;

  // A-fragment k order this code uses (contiguous, same as B): lane holds row m = lane%16;
  //   lanes 0-15: K = 0..15; lanes 16-31: K = 16..31
  const int aKoff = halfSel * 16;
  // B-fragment k order this code uses (same as A): lane holds col n = lane%16;
  //   lanes 0-15: K = 0..15; lanes 16-31: K = 16..31
  const int bKoff = halfSel * 16;

  const bf16* aRow0 = A + (size_t)(mBase + l16) * lda;
  const bf16* aRow1 = aRow0 + (size_t)16 * lda;

  // Stage-copy mapping: B tile is 128 rows x 64 k bf16 = 16KB; 256 threads x
  // 16B per copy instruction -> 4 copies per thread per stage.
  // chunk g = i*256 + tid in [0,1024): row r = g>>3, chunk c = g&7 (8 bf16).
  const bf16* bsrc[4];
  int bdst[4];
  #pragma unroll
  for (int i = 0; i < 4; ++i) {
    const int g = i * 256 + tid;
    const int r = g >> 3;
    const int c = g & 7;
    bsrc[i] = B + (size_t)(nBase0 + r) * ldb + c * 8;
    bdst[i] = r * BPITCH + c * 8;
  }

  // prologue: stage 0
  #pragma unroll
  for (int i = 0; i < 4; ++i) ASYNC_CP16(&Bs[0][bdst[i]], bsrc[i]);

  v8f acc[2][4] = {};
  const int T = K / KT;

  for (int t = 0; t < T; ++t) {
    if (t + 1 < T) {
      const size_t koff = (size_t)(t + 1) * KT;
      bf16* stg = &Bs[(t + 1) & 1][0];
      #pragma unroll
      for (int i = 0; i < 4; ++i) ASYNC_CP16(stg + bdst[i], bsrc[i] + koff);
      WAIT_ASYNC(4);   // stage t's 4 copies retired (in-order), next 4 in flight
    } else {
      WAIT_ASYNC(0);
    }
    __syncthreads();

    const bf16* bs = &Bs[t & 1][0];
    #pragma unroll
    for (int ks = 0; ks < 2; ++ks) {
      const int k0 = t * KT + ks * 32;
      // Issue ALL operand loads for this k-step first (4 global + 8 ds),
      // into independent registers, so the compiler can clause them and
      // drain with incremental waits instead of one wait per WMMA pair.
      AFrag a0, a1, bfrag[4];
      a0.h[0] = *(const v8bf*)(aRow0 + k0 + aKoff);
      a0.h[1] = *(const v8bf*)(aRow0 + k0 + 8 + aKoff);
      a1.h[0] = *(const v8bf*)(aRow1 + k0 + aKoff);
      a1.h[1] = *(const v8bf*)(aRow1 + k0 + 8 + aKoff);
      #pragma unroll
      for (int j = 0; j < 4; ++j) {
        const bf16* bp = bs + (nWave + j * 16 + l16) * BPITCH + ks * 32 + bKoff;
        bfrag[j].h[0] = *(const v8bf*)(bp);
        bfrag[j].h[1] = *(const v8bf*)(bp + 8);
      }
      #pragma unroll
      for (int j = 0; j < 4; ++j) {
        acc[0][j] = __builtin_amdgcn_wmma_f32_16x16x32_bf16(
            false, a0.v, false, bfrag[j].v, (short)0, acc[0][j], false, false);
        acc[1][j] = __builtin_amdgcn_wmma_f32_16x16x32_bf16(
            false, a1.v, false, bfrag[j].v, (short)0, acc[1][j], false, false);
      }
    }
    __syncthreads();   // stage buffer (t&1) free for overwrite at t+2's issue
  }

  // C/D layout: VGPR r -> M = r + (lane>=16 ? 8 : 0), N = lane%16
  const int rowOff = halfSel * 8;
  #pragma unroll
  for (int mi = 0; mi < 2; ++mi) {
    #pragma unroll
    for (int nj = 0; nj < 4; ++nj) {
      const int n = nBase0 + nWave + nj * 16 + l16;
      const float bv = HAS_BIAS ? bias[n] : 0.0f;
      #pragma unroll
      for (int r = 0; r < 8; ++r) {
        const int m = mBase + mi * 16 + rowOff + r;
        const float val = acc[mi][nj][r] * scale + bv;
        if (OUT_BF16) ((bf16*)Cout)[(size_t)m * ldc + n] = (bf16)val;
        else          ((float*)Cout)[(size_t)m * ldc + n] = val;
      }
    }
  }
}

// ---------------------------------------------------------------------------
// Row softmax over S[D,D]; writes fp32 attn (tuple output) + bf16 copy for G3.
// One 256-thread block per row.
// ---------------------------------------------------------------------------
__global__ __launch_bounds__(256) void k_softmax_row(
    const float* __restrict__ S,
    float* __restrict__ attnF,
    bf16*  __restrict__ attnB, int D)
{
  __shared__ float red[256];
  const int row = blockIdx.x;
  const int tid = threadIdx.x;
  const float* s = S + (size_t)row * D;

  float mx = -3.0e38f;
  for (int c = tid; c < D; c += 256) mx = fmaxf(mx, s[c]);
  red[tid] = mx; __syncthreads();
  for (int w = 128; w > 0; w >>= 1) {
    if (tid < w) red[tid] = fmaxf(red[tid], red[tid + w]);
    __syncthreads();
  }
  mx = red[0]; __syncthreads();

  float sum = 0.0f;
  for (int c = tid; c < D; c += 256) sum += __expf(s[c] - mx);
  red[tid] = sum; __syncthreads();
  for (int w = 128; w > 0; w >>= 1) {
    if (tid < w) red[tid] += red[tid + w];
    __syncthreads();
  }
  const float inv = 1.0f / red[0];

  for (int c = tid; c < D; c += 256) {
    const float v = __expf(s[c] - mx) * inv;
    attnF[(size_t)row * D + c] = v;
    attnB[(size_t)row * D + c] = (bf16)v;
  }
}

// ---------------------------------------------------------------------------
// Host-side orchestration
// ---------------------------------------------------------------------------
extern "C" void kernel_launch(void* const* d_in, const int* in_sizes, int n_in,
                              void* d_out, int out_size, void* d_ws, size_t ws_size,
                              hipStream_t stream) {
  (void)in_sizes; (void)n_in; (void)out_size; (void)ws_size;

  const float* x    = (const float*)d_in[0];   // [N, D]
  const float* Wqkv = (const float*)d_in[1];   // [3D, D]
  const float* bqkv = (const float*)d_in[2];   // [3D]
  const float* Wout = (const float*)d_in[3];   // [D, D]
  const float* bout = (const float*)d_in[4];   // [D]

  float* outY    = (float*)d_out;                       // [N, D]
  float* attnOut = outY + (size_t)DIM_N * DIM_D;        // [D, D]

  // --- workspace carve (bf16 copies + intermediates), ~530 MB total ---
  char* p = (char*)d_ws;
  bf16* xb    = (bf16*)p;  p += (size_t)DIM_N  * DIM_D  * 2;  // 64 MB
  bf16* wqb   = (bf16*)p;  p += (size_t)DIM_3D * DIM_D  * 2;  // 24 MB
  bf16* wob   = (bf16*)p;  p += (size_t)DIM_D  * DIM_D  * 2;  //  8 MB
  bf16* qkvb  = (bf16*)p;  p += (size_t)DIM_N  * DIM_3D * 2;  // 192 MB
  bf16* qT    = (bf16*)p;  p += (size_t)DIM_D  * DIM_N  * 2;  // 64 MB
  bf16* kT    = (bf16*)p;  p += (size_t)DIM_D  * DIM_N  * 2;  // 64 MB
  float* Sbuf = (float*)p; p += (size_t)DIM_D  * DIM_D  * 4;  // 16 MB
  bf16* attnB = (bf16*)p;  p += (size_t)DIM_D  * DIM_D  * 2;  //  8 MB
  bf16* out2b = (bf16*)p;  p += (size_t)DIM_D  * DIM_N  * 2;  // 64 MB

  // --- fp32 -> bf16 conversions ---
  {
    int n = DIM_N * DIM_D;
    k_cvt_f32_bf16<<<(n + 255) / 256, 256, 0, stream>>>(x, xb, n);
  }
  {
    int n = DIM_3D * DIM_D;
    k_cvt_f32_bf16<<<(n + 255) / 256, 256, 0, stream>>>(Wqkv, wqb, n);
  }
  {
    int n = DIM_D * DIM_D;
    k_cvt_f32_bf16<<<(n + 255) / 256, 256, 0, stream>>>(Wout, wob, n);
  }

  // --- G1: qkv[N,3D] = x @ Wqkv^T + bqkv  (bf16 out) ---
  {
    dim3 grid(DIM_3D / 128, DIM_N / 128);
    k_gemm_abt<true, true><<<grid, 256, 0, stream>>>(
        xb, DIM_D, wqb, DIM_D, bqkv, qkvb, DIM_3D, DIM_D, 1.0f);
  }

  // --- transpose q, k slices of qkv -> qT[D,N], kT[D,N] ---
  {
    dim3 grid(DIM_D / 32, DIM_N / 32);
    k_transpose_bf16<<<grid, 256, 0, stream>>>(qkvb,         DIM_3D, qT, DIM_N);
    k_transpose_bf16<<<grid, 256, 0, stream>>>(qkvb + DIM_D, DIM_3D, kT, DIM_N);
  }

  // --- G2: S[D,D] = (qT @ kT^T) / sqrt(N)  (fp32 out) ---
  {
    dim3 grid(DIM_D / 128, DIM_D / 128);
    const float invs = 1.0f / sqrtf((float)DIM_N);
    k_gemm_abt<false, false><<<grid, 256, 0, stream>>>(
        qT, DIM_N, kT, DIM_N, nullptr, Sbuf, DIM_D, DIM_N, invs);
  }

  // --- softmax rows: attn fp32 (tuple output #2) + bf16 copy ---
  k_softmax_row<<<DIM_D, 256, 0, stream>>>(Sbuf, attnOut, attnB, DIM_D);

  // --- G3: out2[D,N] = attn @ v^T, v = qkv[:, 2D:3D]  (bf16 out) ---
  {
    dim3 grid(DIM_N / 128, DIM_D / 128);
    k_gemm_abt<true, false><<<grid, 256, 0, stream>>>(
        attnB, DIM_D, qkvb + 2 * DIM_D, DIM_3D, nullptr, out2b, DIM_N, DIM_D, 1.0f);
  }

  // --- G4: Y[N,D] = out2(flat as [N,D]) @ Wout^T + bout  (fp32 out) ---
  {
    dim3 grid(DIM_D / 128, DIM_N / 128);
    k_gemm_abt<false, true><<<grid, 256, 0, stream>>>(
        out2b, DIM_D, wob, DIM_D, bout, outY, DIM_D, DIM_D, 1.0f);
  }
}
